// DynamicGNNv2_74036646248566
// MI455X (gfx1250) — matrix operand order repaired and checked by reading, not compiled, not run on hardware
//
#include <hip/hip_runtime.h>
#include <stdint.h>

typedef __attribute__((ext_vector_type(16))) _Float16 v16h;
typedef __attribute__((ext_vector_type(8)))  float    v8f;

#define GNN_D 32
#define GNN_EPS 1e-5f

// ---------------------------------------------------------------- utilities

__global__ void zero_f32_kernel(float* __restrict__ p, int n) {
  int i = blockIdx.x * blockDim.x + threadIdx.x;
  if (i < n) p[i] = 0.0f;
}

__global__ void degree_kernel(const long long* __restrict__ dst,
                              float* __restrict__ deg, int E) {
  int e = blockIdx.x * blockDim.x + threadIdx.x;
  if (e < E) atomicAdd(&deg[(int)dst[e]], 1.0f);
}

__global__ void invdeg_kernel(float* __restrict__ deg, int n) {
  int i = blockIdx.x * blockDim.x + threadIdx.x;
  if (i < n) deg[i] = 1.0f / fmaxf(deg[i], 1.0f);
}

// h[n][d] = sum_k x[n][k] * w_in[d][k] + b_in[d]   (K=5: plain FMA, trivial)
__global__ void input_proj_kernel(const float* __restrict__ x,
                                  const float* __restrict__ w_in,
                                  const float* __restrict__ b_in,
                                  float* __restrict__ h, int n) {
  int t = blockIdx.x * blockDim.x + threadIdx.x;
  int node = t >> 5;
  int d = t & 31;
  if (node >= n) return;
  const float* xr = x + (size_t)node * 5;
  const float* wr = w_in + (size_t)d * 5;
  float acc = b_in[d];
#pragma unroll
  for (int k = 0; k < 5; ++k) acc = fmaf(xr[k], wr[k], acc);
  h[(size_t)node * GNN_D + d] = acc;
}

// One wave per 32 edges. Each lane loads its own src/dst pair (coalesced
// 256B), then an unrolled loop broadcasts each index to an SGPR via
// v_readlane so the 128B row gather + 128B f32-atomic row scatter use
// scalar-base addressing. This is the bandwidth-dominant kernel.
__global__ void edge_scatter_kernel(const float* __restrict__ h,
                                    const long long* __restrict__ ei,
                                    float* __restrict__ agg, int E) {
  const int lane = threadIdx.x & 31;
  const int wid  = (int)((blockIdx.x * blockDim.x + threadIdx.x) >> 5);
  const int base = wid * 32;
  if (base >= E) return;                       // wave-uniform
  const int eidx = min(base + lane, E - 1);
  const int s32  = (int)ei[eidx];
  const int d32  = (int)ei[(size_t)E + eidx];
  const int cnt  = min(32, E - base);          // wave-uniform
#pragma unroll
  for (int e = 0; e < 32; ++e) {
    if (e < cnt) {                             // uniform branch
      int s = __builtin_amdgcn_readlane(s32, e);
      int d = __builtin_amdgcn_readlane(d32, e);
      float v = h[(size_t)s * GNN_D + lane];
      atomicAdd(&agg[(size_t)d * GNN_D + lane], v);
    }
  }
}

// ------------------------------------------------------- fused WMMA layer

__device__ __forceinline__ v16h pack16(float4 a, float4 b, float4 c, float4 d,
                                       float s) {
  v16h v;
  v[0]  = (_Float16)(a.x * s); v[1]  = (_Float16)(a.y * s);
  v[2]  = (_Float16)(a.z * s); v[3]  = (_Float16)(a.w * s);
  v[4]  = (_Float16)(b.x * s); v[5]  = (_Float16)(b.y * s);
  v[6]  = (_Float16)(b.z * s); v[7]  = (_Float16)(b.w * s);
  v[8]  = (_Float16)(c.x * s); v[9]  = (_Float16)(c.y * s);
  v[10] = (_Float16)(c.z * s); v[11] = (_Float16)(c.w * s);
  v[12] = (_Float16)(d.x * s); v[13] = (_Float16)(d.y * s);
  v[14] = (_Float16)(d.z * s); v[15] = (_Float16)(d.w * s);
  return v;
}

// h_out = LayerNorm(ReLU( (agg*inv_deg) @ w_l^T + b_l + h_in @ w_r^T ))
// Grid-stride loop over 16-node tiles; weights/affine params loaded ONCE per
// wave and held in VGPRs; 4x v_wmma_f32_16x16x32_f16 per tile.
__global__ void sage_wmma_kernel(const float* __restrict__ h_in,
                                 const float* __restrict__ agg,
                                 const float* __restrict__ inv_deg,
                                 const float* __restrict__ w_l,
                                 const float* __restrict__ b_l,
                                 const float* __restrict__ w_r,
                                 const float* __restrict__ gamma,
                                 const float* __restrict__ beta,
                                 float* __restrict__ h_out, int n) {
  const int lane  = threadIdx.x & 31;
  const int wave  = threadIdx.x >> 5;
  const int mloc  = lane & 15;               // M row for A / N col for B,C
  const int khalf = lane >> 4;               // which K-half this lane holds
  const int ntiles = (n + 15) >> 4;
  const int wstride = gridDim.x * 8;         // total waves
  int tile = blockIdx.x * 8 + wave;
  if (tile >= ntiles) return;                // wave-uniform

  // ---- B matrices (held across tiles): Wt slice 32x16; lane = output
  // column, khalf = K-half.  B[k][ng] = W[ng][k] -> 16 contiguous floats of
  // W row ng starting at 16*khalf.
  v16h Bl[2], Br[2];
#pragma unroll
  for (int oh = 0; oh < 2; ++oh) {
    const int ng = oh * 16 + mloc;
    const float4* wl = (const float4*)(w_l + (size_t)ng * GNN_D) + 4 * khalf;
    const float4* wr = (const float4*)(w_r + (size_t)ng * GNN_D) + 4 * khalf;
    Bl[oh] = pack16(wl[0], wl[1], wl[2], wl[3], 1.0f);
    Br[oh] = pack16(wr[0], wr[1], wr[2], wr[3], 1.0f);
  }
  const float bias0 = b_l[mloc],  bias1 = b_l[16 + mloc];
  const float g0 = gamma[mloc],   g1 = gamma[16 + mloc];
  const float e0 = beta[mloc],    e1 = beta[16 + mloc];

  for (; tile < ntiles; tile += wstride) {
    const int node = min(tile * 16 + mloc, n - 1);  // clamp tail, stores guarded

    // ---- A matrices: 16x32 f16, contiguous K layout (lanes 0-15: K 0..15;
    //      lanes 16-31: K 16..31), matching B.
    const float4* hr = (const float4*)(h_in + (size_t)node * GNN_D);
    const float4* ar = (const float4*)(agg  + (size_t)node * GNN_D);
    const float   sc = inv_deg[node];
    v16h Ah = pack16(hr[4 * khalf], hr[4 * khalf + 1],
                     hr[4 * khalf + 2], hr[4 * khalf + 3], 1.0f);
    v16h Aa = pack16(ar[4 * khalf], ar[4 * khalf + 1],
                     ar[4 * khalf + 2], ar[4 * khalf + 3], sc);

    // ---- accumulators preloaded with lin_l bias (bias depends only on N col)
    v8f acc0, acc1;
#pragma unroll
    for (int r = 0; r < 8; ++r) { acc0[r] = bias0; acc1[r] = bias1; }

    acc0 = __builtin_amdgcn_wmma_f32_16x16x32_f16(false, Aa, false, Bl[0],
                                                  (short)0, acc0, false, false);
    acc0 = __builtin_amdgcn_wmma_f32_16x16x32_f16(false, Ah, false, Br[0],
                                                  (short)0, acc0, false, false);
    acc1 = __builtin_amdgcn_wmma_f32_16x16x32_f16(false, Aa, false, Bl[1],
                                                  (short)0, acc1, false, false);
    acc1 = __builtin_amdgcn_wmma_f32_16x16x32_f16(false, Ah, false, Br[1],
                                                  (short)0, acc1, false, false);

    // ---- fused ReLU + LayerNorm(32) + affine + coalesced row stores.
    // C layout: lane holds column (oh*16+mloc); VGPR r holds row r + 8*khalf.
    // Row reduction = butterfly over 16 lanes sharing khalf (masks 1..8).
#pragma unroll
    for (int r = 0; r < 8; ++r) {
      float v0 = fmaxf(acc0[r], 0.0f);
      float v1 = fmaxf(acc1[r], 0.0f);
      float s  = v0 + v1;
      float ss = v0 * v0 + v1 * v1;
#pragma unroll
      for (int m = 1; m < 16; m <<= 1) {
        s  += __shfl_xor(s,  m, 32);
        ss += __shfl_xor(ss, m, 32);
      }
      float mean = s * (1.0f / 32.0f);
      float var  = ss * (1.0f / 32.0f) - mean * mean;
      float rstd = rsqrtf(var + GNN_EPS);
      int row = tile * 16 + r + 8 * khalf;
      if (row < n) {
        float* o = h_out + (size_t)row * GNN_D;
        o[mloc]      = (v0 - mean) * rstd * g0 + e0;
        o[16 + mloc] = (v1 - mean) * rstd * g1 + e1;
      }
    }
  }
}

// ---------------------------------------------------------------- launcher

extern "C" void kernel_launch(void* const* d_in, const int* in_sizes, int n_in,
                              void* d_out, int out_size, void* d_ws,
                              size_t ws_size, hipStream_t stream) {
  const float*     x     = (const float*)d_in[0];
  const long long* ei    = (const long long*)d_in[1];   // [2,E] int64
  const float*     w_in  = (const float*)d_in[2];
  const float*     b_in  = (const float*)d_in[3];
  const float*     w_l   = (const float*)d_in[4];       // [L,32,32]
  const float*     b_l   = (const float*)d_in[5];       // [L,32]
  const float*     w_r   = (const float*)d_in[6];       // [L,32,32]
  const float*     gamma = (const float*)d_in[7];
  const float*     beta  = (const float*)d_in[8];

  const int n = in_sizes[0] / 5;       // N nodes
  const int E = in_sizes[1] / 2;       // edges
  const int L = in_sizes[5] / GNN_D;   // layers

  float* ws   = (float*)d_ws;
  float* h_a  = ws;                              // N*32
  float* h_b  = ws + (size_t)n * GNN_D;          // N*32
  float* agg  = ws + (size_t)2 * n * GNN_D;      // N*32
  float* deg  = ws + (size_t)3 * n * GNN_D;      // N (becomes inv_deg)

  const int nd = n * GNN_D;

  // degree -> inv_deg (once)
  zero_f32_kernel<<<(n + 255) / 256, 256, 0, stream>>>(deg, n);
  degree_kernel<<<(E + 255) / 256, 256, 0, stream>>>(ei + E, deg, E);
  invdeg_kernel<<<(n + 255) / 256, 256, 0, stream>>>(deg, n);

  // input projection
  input_proj_kernel<<<(n * 32 + 255) / 256, 256, 0, stream>>>(x, w_in, b_in,
                                                              h_a, n);

  float* hin = h_a;
  float* hout = h_b;
  const int ntiles = (n + 15) / 16;
  // ~2 tiles per wave so per-wave weight loads amortize; 8 waves per block.
  const int gemm_blocks = (ntiles + 15) / 16;
  const int scat_blocks = ((E + 31) / 32 + 7) / 8;   // 8 waves/block, 32 edges/wave
  for (int l = 0; l < L; ++l) {
    zero_f32_kernel<<<(nd + 255) / 256, 256, 0, stream>>>(agg, nd);
    edge_scatter_kernel<<<scat_blocks, 256, 0, stream>>>(hin, ei, agg, E);
    float* out = (l == L - 1) ? (float*)d_out : hout;
    sage_wmma_kernel<<<gemm_blocks, 256, 0, stream>>>(
        hin, agg, deg, w_l + (size_t)l * GNN_D * GNN_D, b_l + (size_t)l * GNN_D,
        w_r + (size_t)l * GNN_D * GNN_D, gamma, beta, out, n);
    float* t = hin; hin = hout; hout = t;
    hin = out;
  }
}
